// MaskedAttentionLayer_59390807769675
// MI455X (gfx1250) — matrix operand order repaired and checked by reading, not compiled, not run on hardware
//
#include <hip/hip_runtime.h>
#include <stdint.h>

#define EMBED 1024
#define SEQ   2048
#define BATCH 4
#define LDT   40          // padded LDS K-stride (bf16 elems): 32 data + 8 pad

typedef __attribute__((ext_vector_type(16))) __bf16 v16bf;
typedef __attribute__((ext_vector_type(8)))  float  v8f;
typedef __attribute__((ext_vector_type(4)))  int    v4i;

// ---------------- CDNA5 async global->LDS path (guarded) ----------------

#if __has_builtin(__builtin_amdgcn_global_load_async_to_lds_b128)
#define HAS_ASYNC_LDS 1
typedef __attribute__((address_space(1))) v4i* gvec_p;
typedef __attribute__((address_space(3))) v4i* lvec_p;
__device__ __forceinline__ void async_copy16(const void* g, void* l) {
  __builtin_amdgcn_global_load_async_to_lds_b128((gvec_p)g, (lvec_p)l, 0, 0);
}
#if __has_builtin(__builtin_amdgcn_s_wait_asynccnt)
#define WAIT_ASYNC() __builtin_amdgcn_s_wait_asynccnt(0)
#else
#define WAIT_ASYNC() asm volatile("s_wait_asynccnt 0x0" ::: "memory")
#endif
#else
#define HAS_ASYNC_LDS 0
#define WAIT_ASYNC()
#endif

// ---------------- helpers ----------------

__device__ __forceinline__ __bf16 f32_to_bf16(float f) {
  union { float f; uint32_t u; } c; c.f = f;
  uint32_t u = c.u;
  u += 0x7FFFu + ((u >> 16) & 1u);           // round-to-nearest-even
  unsigned short s = (unsigned short)(u >> 16);
  __bf16 out;
  __builtin_memcpy(&out, &s, sizeof(out));
  return out;
}

// A fragment: 16x32 bf16, row-major source (lda in elements).
// ISA layout: lanes 0-15 -> M=lane, K in {0..7,16..23}; lanes 16-31 -> K in {8..15,24..31}.
__device__ __forceinline__ void load_a_frag(v16bf& a, const __bf16* base, int lda, int lane) {
  const int m = lane & 15;
  const int g = lane >> 4;
  const __bf16* row = base + m * lda;
#pragma unroll
  for (int v = 0; v < 8; ++v) {
    const int k = ((v & 4) << 2) + g * 8 + ((v & 3) << 1);
    a[2 * v]     = row[k];
    a[2 * v + 1] = row[k + 1];
  }
}

// B fragment: 32x16 bf16 (KxN). Source laid out [N][K] with K contiguous (ldb in elems).
// ISA layout: lane -> N = lane&15; lanes 0-15 hold K in {0..7,16..23}, lanes 16-31 hold K in {8..15,24..31}.
__device__ __forceinline__ void load_b_frag(v16bf& b, const __bf16* base, int ldb, int lane) {
  const int n  = lane & 15;
  const int kb = (lane >> 4) << 3;
  const __bf16* p = base + n * ldb + kb;
#pragma unroll
  for (int i = 0; i < 16; ++i) b[i] = p[(i & 7) + ((i >> 3) << 4)];
}

// Stage one 128x32 A tile and one 128x32 B tile (both K-contiguous) into LDS.
// 512 16-byte chunks per matrix, 256 threads -> 2 chunks each.
__device__ __forceinline__ void stage_pair(__bf16 (*As)[LDT], __bf16 (*Bs)[LDT],
                                           const __bf16* __restrict__ A, int lda,
                                           const __bf16* __restrict__ B, int ldb,
                                           int k0, int t) {
#pragma unroll
  for (int rep = 0; rep < 2; ++rep) {
    const int c = t + rep * 256;
    const int row = c >> 2, kc = (c & 3) << 3;   // kc in elements (8 bf16 = 16B)
#if HAS_ASYNC_LDS
    async_copy16(A + (size_t)row * lda + k0 + kc, &As[row][kc]);
    async_copy16(B + (size_t)row * ldb + k0 + kc, &Bs[row][kc]);
#else
    *reinterpret_cast<uint4*>(&As[row][kc]) =
        *reinterpret_cast<const uint4*>(A + (size_t)row * lda + k0 + kc);
    *reinterpret_cast<uint4*>(&Bs[row][kc]) =
        *reinterpret_cast<const uint4*>(B + (size_t)row * ldb + k0 + kc);
#endif
  }
}

// ---------------- GEMM core: 128x128 tile, 8 waves, K-step 32, double-buffered ----------------
// A: [M][K] row-major (K contiguous).  B: [N][K] (K contiguous).
__device__ __forceinline__ void gemm_tile(const __bf16* __restrict__ A, int lda,
                                          const __bf16* __restrict__ Bm, int ldb,
                                          int kLen, v8f acc[2][4]) {
  __shared__ __bf16 As[2][128][LDT];
  __shared__ __bf16 Bs[2][128][LDT];
  const int t    = threadIdx.x;
  const int lane = t & 31;
  const int wave = t >> 5;
  const int wm   = (wave & 3) * 32;   // wave's M sub-origin
  const int wn   = (wave >> 2) * 64;  // wave's N sub-origin

  stage_pair(As[0], Bs[0], A, lda, Bm, ldb, 0, t);
  WAIT_ASYNC();
  __syncthreads();

  for (int k0 = 0; k0 < kLen; k0 += 32) {
    const int buf = (k0 >> 5) & 1;
    if (k0 + 32 < kLen)                       // prefetch next K-slab into other buffer
      stage_pair(As[buf ^ 1], Bs[buf ^ 1], A, lda, Bm, ldb, k0 + 32, t);

    v16bf a[2], bfr[4];
#pragma unroll
    for (int i = 0; i < 2; ++i) load_a_frag(a[i], &As[buf][wm + i * 16][0], LDT, lane);
#pragma unroll
    for (int j = 0; j < 4; ++j) load_b_frag(bfr[j], &Bs[buf][wn + j * 16][0], LDT, lane);
#pragma unroll
    for (int i = 0; i < 2; ++i)
#pragma unroll
      for (int j = 0; j < 4; ++j)
        acc[i][j] = __builtin_amdgcn_wmma_f32_16x16x32_bf16(
            false, a[i], false, bfr[j], (short)0, acc[i][j], false, false);

    WAIT_ASYNC();
    __syncthreads();
  }
}

// ---------------- kernels ----------------

__global__ void cvt_f32_bf16(const float* __restrict__ src, __bf16* __restrict__ dst, int n4) {
  const int i = blockIdx.x * blockDim.x + threadIdx.x;
  if (i < n4) {
    const float4 v = reinterpret_cast<const float4*>(src)[i];
    struct bf16x4 { __bf16 a, b, c, d; } o;
    o.a = f32_to_bf16(v.x); o.b = f32_to_bf16(v.y);
    o.c = f32_to_bf16(v.z); o.d = f32_to_bf16(v.w);
    reinterpret_cast<bf16x4*>(dst)[i] = o;
  }
}

// QKV projection: Y[m][n] = sum_k x[m][k] * W[n][k] + bias[n].
// which==0 (Q): scale by 1/sqrt(E), row-major out.  which==1 (K): row-major out.
// which==2 (V): store TRANSPOSED as Vt[b][e][s] so P@V consumes it K-contiguous.
__global__ __launch_bounds__(256) void qkv_gemm(
    const __bf16* __restrict__ xb, const __bf16* __restrict__ Wall,
    const float* __restrict__ bqp, const float* __restrict__ bkp, const float* __restrict__ bvp,
    __bf16* __restrict__ Qb, __bf16* __restrict__ Kb, __bf16* __restrict__ Vt) {
  const int mt = blockIdx.x, nt = blockIdx.y, which = blockIdx.z;
  const __bf16* W    = Wall + (size_t)which * EMBED * EMBED;
  const float*  bias = which == 0 ? bqp : (which == 1 ? bkp : bvp);
  const float oscale = which == 0 ? 0.03125f : 1.0f;   // fold 1/sqrt(1024) into Q

  v8f acc[2][4] = {};
  gemm_tile(xb + (size_t)(mt * 128) * EMBED, EMBED,
            W  + (size_t)(nt * 128) * EMBED, EMBED, EMBED, acc);

  const int lane = threadIdx.x & 31, wave = threadIdx.x >> 5;
  const int wm = (wave & 3) * 32, wn = (wave >> 2) * 64;
  const int nIdx = lane & 15, g = lane >> 4;

  if (which == 2) {
    // transposed store: per fragment, the 8 accumulator rows are consecutive s values
#pragma unroll
    for (int i = 0; i < 2; ++i)
#pragma unroll
      for (int j = 0; j < 4; ++j) {
        const int e     = nt * 128 + wn + j * 16 + nIdx;
        const int mbase = mt * 128 + wm + i * 16 + 8 * g;   // global row, multiple of 8
        const int bb    = mbase >> 11;                      // / SEQ
        const int s0    = mbase & (SEQ - 1);
        __bf16 tmp[8];
#pragma unroll
        for (int v = 0; v < 8; ++v) tmp[v] = f32_to_bf16(acc[i][j][v] + bias[e]);
        *reinterpret_cast<uint4*>(&Vt[(size_t)bb * EMBED * SEQ + (size_t)e * SEQ + s0]) =
            *reinterpret_cast<const uint4*>(tmp);
      }
  } else {
    __bf16* out = which == 0 ? Qb : Kb;
#pragma unroll
    for (int i = 0; i < 2; ++i)
#pragma unroll
      for (int j = 0; j < 4; ++j)
#pragma unroll
        for (int v = 0; v < 8; ++v) {
          const int m = mt * 128 + wm + i * 16 + v + 8 * g;
          const int n = nt * 128 + wn + j * 16 + nIdx;
          out[(size_t)m * EMBED + n] = f32_to_bf16((acc[i][j][v] + bias[n]) * oscale);
        }
  }
}

// scores[b][q][k] = Q[b][q] . K[b][k]  (scale already folded into Q), causal-masked.
__global__ __launch_bounds__(256) void scores_gemm(
    const __bf16* __restrict__ Qb, const __bf16* __restrict__ Kb, float* __restrict__ Sb) {
  const int kt = blockIdx.x, qt = blockIdx.y, b = blockIdx.z;
  const int m0 = qt * 128, n0 = kt * 128;
  float* outb = Sb + (size_t)b * SEQ * SEQ;

  if (n0 > m0 + 127) {                 // tile fully above diagonal: mask fill only
    const float4 fill = make_float4(-1e30f, -1e30f, -1e30f, -1e30f);
    for (int c = threadIdx.x; c < 128 * 32; c += 256) {
      const int r = c >> 5, cc = (c & 31) << 2;
      *reinterpret_cast<float4*>(&outb[(size_t)(m0 + r) * SEQ + n0 + cc]) = fill;
    }
    return;
  }

  v8f acc[2][4] = {};
  gemm_tile(Qb + (size_t)b * SEQ * EMBED + (size_t)m0 * EMBED, EMBED,
            Kb + (size_t)b * SEQ * EMBED + (size_t)n0 * EMBED, EMBED, EMBED, acc);

  const int lane = threadIdx.x & 31, wave = threadIdx.x >> 5;
  const int wm = (wave & 3) * 32, wn = (wave >> 2) * 64;
  const int nIdx = lane & 15, g = lane >> 4;
#pragma unroll
  for (int i = 0; i < 2; ++i)
#pragma unroll
    for (int j = 0; j < 4; ++j)
#pragma unroll
      for (int v = 0; v < 8; ++v) {
        const int q  = m0 + wm + i * 16 + v + 8 * g;
        const int kk = n0 + wn + j * 16 + nIdx;
        outb[(size_t)q * SEQ + kk] = (kk <= q) ? acc[i][j][v] : -1e30f;
      }
}

// Row softmax over f32 scores; writes bf16 probabilities IN PLACE over the f32 row
// (bf16 row occupies first half of each 8KB f32 row; row stride stays 2*SEQ bf16 elems).
__global__ __launch_bounds__(256) void softmax_rows(float* __restrict__ Sb) {
  __shared__ float rowbuf[SEQ];
  __shared__ float red[256];
  const int t = threadIdx.x;
  float* src = Sb + (size_t)blockIdx.x * SEQ;

  float lmax = -1e30f;
  for (int i = t; i < SEQ; i += 256) {
    const float v = src[i];
    rowbuf[i] = v;
    lmax = fmaxf(lmax, v);
  }
  red[t] = lmax; __syncthreads();
  for (int s = 128; s > 0; s >>= 1) { if (t < s) red[t] = fmaxf(red[t], red[t + s]); __syncthreads(); }
  const float m = red[0]; __syncthreads();

  float lsum = 0.f;
  for (int i = t; i < SEQ; i += 256) lsum += __expf(rowbuf[i] - m);
  red[t] = lsum; __syncthreads();
  for (int s = 128; s > 0; s >>= 1) { if (t < s) red[t] += red[t + s]; __syncthreads(); }
  const float inv = 1.0f / red[0]; __syncthreads();

  __bf16* dst = reinterpret_cast<__bf16*>(src);
  for (int i = t; i < SEQ; i += 256) dst[i] = f32_to_bf16(__expf(rowbuf[i] - m) * inv);
}

// out[b][q][e] = sum_k P[b][q][k] * Vt[b][e][k]; K-loop truncated at causal bound.
__global__ __launch_bounds__(256) void out_gemm(
    const float* __restrict__ Sb, const __bf16* __restrict__ Vt, float* __restrict__ Out) {
  const int mt = blockIdx.x, nt = blockIdx.y, b = blockIdx.z;
  const __bf16* Pb = reinterpret_cast<const __bf16*>(Sb + (size_t)b * SEQ * SEQ);
  const int kLen = mt * 128 + 128;     // keys beyond q-tile have exactly zero prob

  v8f acc[2][4] = {};
  gemm_tile(Pb + (size_t)(mt * 128) * (2 * SEQ), 2 * SEQ,
            Vt + (size_t)b * EMBED * SEQ + (size_t)(nt * 128) * SEQ, SEQ, kLen, acc);

  const int lane = threadIdx.x & 31, wave = threadIdx.x >> 5;
  const int wm = (wave & 3) * 32, wn = (wave >> 2) * 64;
  const int nIdx = lane & 15, g = lane >> 4;
  float* outb = Out + (size_t)b * SEQ * EMBED;
#pragma unroll
  for (int i = 0; i < 2; ++i)
#pragma unroll
    for (int j = 0; j < 4; ++j)
#pragma unroll
      for (int v = 0; v < 8; ++v) {
        const int q = mt * 128 + wm + i * 16 + v + 8 * g;
        const int e = nt * 128 + wn + j * 16 + nIdx;
        outb[(size_t)q * EMBED + e] = acc[i][j][v];
      }
}

// ---------------- launch ----------------

extern "C" void kernel_launch(void* const* d_in, const int* in_sizes, int n_in,
                              void* d_out, int out_size, void* d_ws, size_t ws_size,
                              hipStream_t stream) {
  (void)in_sizes; (void)n_in; (void)out_size; (void)ws_size;
  const float* x  = (const float*)d_in[0];
  const float* Wq = (const float*)d_in[1];
  const float* bq = (const float*)d_in[2];
  const float* Wk = (const float*)d_in[3];
  const float* bk = (const float*)d_in[4];
  const float* Wv = (const float*)d_in[5];
  const float* bv = (const float*)d_in[6];
  float* out = (float*)d_out;

  char* ws = (char*)d_ws;
  const size_t XN = (size_t)BATCH * SEQ * EMBED;       // 8,388,608 elems
  const size_t WN = (size_t)EMBED * EMBED;             // 1,048,576 elems
  size_t off = 0;
  __bf16* xb = (__bf16*)(ws + off); off += XN * 2;         // 16 MB
  __bf16* Wb = (__bf16*)(ws + off); off += 3 * WN * 2;     //  6 MB
  __bf16* Qb = (__bf16*)(ws + off); off += XN * 2;         // 16 MB
  __bf16* Kb = (__bf16*)(ws + off); off += XN * 2;         // 16 MB
  __bf16* Vt = (__bf16*)(ws + off); off += XN * 2;         // 16 MB (transposed V)
  float*  Sb = (float*)(ws + off);                          // 64 MB (f32 scores -> bf16 probs in place)

  // 1) convert inputs to bf16
  cvt_f32_bf16<<<(int)(XN / 4 + 255) / 256, 256, 0, stream>>>(x,  xb, (int)(XN / 4));
  cvt_f32_bf16<<<(int)(WN / 4 + 255) / 256, 256, 0, stream>>>(Wq, Wb,           (int)(WN / 4));
  cvt_f32_bf16<<<(int)(WN / 4 + 255) / 256, 256, 0, stream>>>(Wk, Wb + WN,      (int)(WN / 4));
  cvt_f32_bf16<<<(int)(WN / 4 + 255) / 256, 256, 0, stream>>>(Wv, Wb + 2 * WN,  (int)(WN / 4));

  // 2) Q/K/V projections (M=8192, N=1024, K=1024); V written transposed
  qkv_gemm<<<dim3(64, 8, 3), 256, 0, stream>>>(xb, Wb, bq, bk, bv, Qb, Kb, Vt);

  // 3) causal-masked scores (per batch: 2048x2048, K=1024)
  scores_gemm<<<dim3(16, 16, 4), 256, 0, stream>>>(Qb, Kb, Sb);

  // 4) row softmax, f32 -> bf16 probabilities in place
  softmax_rows<<<BATCH * SEQ, 256, 0, stream>>>(Sb);

  // 5) output = P @ V (per batch: M=2048, N=1024, K<=2048 causal-truncated)
  out_gemm<<<dim3(16, 8, 4), 256, 0, stream>>>(Sb, Vt, out);
}
